// sLSTMBlock_69956427317885
// MI455X (gfx1250) — compile-verified, matrix operand order checked by reading, not run on hardware
//
#include <hip/hip_runtime.h>
#include <math.h>

#define BATCH 8192
#define DIM   1024
#define HIDN  1024
#define NH    8
#define HS    128
#define PDIM  1365
#define PPAD  1376   // PDIM padded to multiple of 16
#define KCONV 4

typedef __attribute__((ext_vector_type(16))) _Float16 v16h;
typedef __attribute__((ext_vector_type(8)))  _Float16 v8h;
typedef __attribute__((ext_vector_type(8)))  float    v8f;

// ---------------------------------------------------------------------------
// WMMA fragment loading.
// A 16x32 f16 layout (ISA 7.12.2): lanes 0-15 -> rows M=0..15 with K={0..7,16..23},
// lanes 16-31 -> rows M=0..15 with K={8..15,24..31}. Element 2v/2v+1 of the v16h
// map to lo/hi halves of VGPR v. B operand: identical pattern on a [N][K]
// (transposed) weight image, so one loader serves both.
// ---------------------------------------------------------------------------
__device__ __forceinline__ v16h load_frag16(const _Float16* p) {
  v8h lo = *reinterpret_cast<const v8h*>(p);        // K chunk base .. base+7
  v8h hi = *reinterpret_cast<const v8h*>(p + 16);   // K chunk base+16 .. base+23
  v16h r;
#pragma unroll
  for (int i = 0; i < 8; ++i) { r[i] = lo[i]; r[8 + i] = hi[i]; }
  return r;
}

// tile00 points at element (m/n = 0, k = k0) of a row-major [16+, ld] f16 image.
// ld and k0 must be multiples of 8 halves (16B alignment for the vector loads).
__device__ __forceinline__ v16h load_tile_mk(const _Float16* tile00, int ld) {
  int lane = threadIdx.x & 31;
  int mn   = lane & 15;
  int koff = ((lane >> 4) & 1) * 8;
  return load_frag16(tile00 + mn * ld + koff);
}

__device__ __forceinline__ v8f wmma_f16(v16h a, v16h b, v8f c) {
  return __builtin_amdgcn_wmma_f32_16x16x32_f16(false, a, false, b,
                                                (short)0, c, false, false);
}

// ---------------------------------------------------------------------------
// Kernel 1: per-row RMSNorm + causal conv (K=4, left pad) + SiLU.
// Emits f16 images of raw x (z/o gates), x_conv (i/f gates) and h_prev.
// One 256-thread block per row (1024 elements, 4 per thread).
// ---------------------------------------------------------------------------
__global__ __launch_bounds__(256) void k_prep(
    const float* __restrict__ x, const float* __restrict__ h_prev,
    const float* __restrict__ rms_w, const float* __restrict__ conv_w,
    const float* __restrict__ conv_b,
    _Float16* __restrict__ xh, _Float16* __restrict__ xch,
    _Float16* __restrict__ hh) {
  __shared__ float xn[DIM + KCONV - 1];
  __shared__ float red[256];
  const int row = blockIdx.x;
  const float* xr = x + (size_t)row * DIM;

  float vx[4];
  float ss = 0.f;
#pragma unroll
  for (int i = 0; i < 4; ++i) {
    int j = threadIdx.x + 256 * i;
    vx[i] = xr[j];
    ss += vx[i] * vx[i];
  }
  red[threadIdx.x] = ss;
  __syncthreads();
  for (int s = 128; s > 0; s >>= 1) {
    if (threadIdx.x < s) red[threadIdx.x] += red[threadIdx.x + s];
    __syncthreads();
  }
  const float inv = rsqrtf(red[0] * (1.0f / DIM) + 1e-6f);

  if (threadIdx.x < KCONV - 1) xn[threadIdx.x] = 0.f;  // left zero pad
#pragma unroll
  for (int i = 0; i < 4; ++i) {
    int j = threadIdx.x + 256 * i;
    xn[KCONV - 1 + j] = vx[i] * inv * rms_w[j];
    xh[(size_t)row * DIM + j] = (_Float16)vx[i];
    hh[(size_t)row * DIM + j] = (_Float16)h_prev[(size_t)row * DIM + j];
  }
  __syncthreads();

  const float w0 = conv_w[0], w1 = conv_w[1], w2 = conv_w[2], w3 = conv_w[3];
  const float cb = conv_b[0];
#pragma unroll
  for (int i = 0; i < 4; ++i) {
    int j = threadIdx.x + 256 * i;
    float v = cb + w0 * xn[j] + w1 * xn[j + 1] + w2 * xn[j + 2] + w3 * xn[j + 3];
    v = v / (1.f + expf(-v));  // SiLU
    xch[(size_t)row * DIM + j] = (_Float16)v;
  }
}

// ---------------------------------------------------------------------------
// Kernel 2: weight convert + transpose fp32 [b][K][N] -> f16 [b][Np][Kp],
// zero padded. Gives B operands the same contiguous-K layout as A operands.
// ---------------------------------------------------------------------------
__global__ __launch_bounds__(256) void k_wt(
    const float* __restrict__ src, _Float16* __restrict__ dst,
    int K, int N, int Kp, int Np) {
  const size_t b   = blockIdx.z;
  const size_t idx = (size_t)blockIdx.x * 256 + threadIdx.x;
  const size_t tot = (size_t)Np * Kp;
  if (idx >= tot) return;
  const int n = (int)(idx / Kp);
  const int k = (int)(idx % Kp);
  const float v = (n < N && k < K) ? src[b * ((size_t)K * N) + (size_t)k * N + n]
                                   : 0.f;
  dst[b * tot + (size_t)n * Kp + k] = (_Float16)v;
}

// ---------------------------------------------------------------------------
// Kernel 3: fused block-diagonal gate GEMMs + sLSTM cell + per-head GroupNorm.
// Block = (32, 8): wave y == head. One block handles a 16-row batch tile.
// Per wave: A fragments for x / x_conv / h held resident (12 v16h);
// 8 N-tiles x 4 gates x (4 W + 4 R) K-steps = 256 v_wmma per wave.
// ---------------------------------------------------------------------------
__global__ __launch_bounds__(256) void k_cell(
    const _Float16* __restrict__ xh,  const _Float16* __restrict__ xch,
    const _Float16* __restrict__ hh,
    const _Float16* __restrict__ wzT, const _Float16* __restrict__ wiT,
    const _Float16* __restrict__ wfT, const _Float16* __restrict__ woT,
    const _Float16* __restrict__ rzT, const _Float16* __restrict__ riT,
    const _Float16* __restrict__ rfT, const _Float16* __restrict__ roT,
    const float* __restrict__ wz_b, const float* __restrict__ wi_b,
    const float* __restrict__ wf_b, const float* __restrict__ wo_b,
    const float* __restrict__ rz_b, const float* __restrict__ ri_b,
    const float* __restrict__ rf_b, const float* __restrict__ ro_b,
    const float* __restrict__ c_prev, const float* __restrict__ n_prev,
    const float* __restrict__ m_prev,
    const float* __restrict__ gn_w, const float* __restrict__ gn_b,
    float* __restrict__ out_h, float* __restrict__ out_c,
    float* __restrict__ out_n, float* __restrict__ out_m,
    _Float16* __restrict__ onorm) {
  __shared__ _Float16 ht_s[16][HIDN];       // 32 KB staging for GroupNorm
  __shared__ float mu_s[NH][16], iv_s[NH][16];

  const int lane = threadIdx.x;
  const int h    = threadIdx.y;
  const int row0 = blockIdx.x * 16;
  const int rsel  = lane >> 4;
  const int nlane = lane & 15;

  // Resident A fragments for this head's 128-wide input slice.
  v16h ax[4], ac[4], ah[4];
  const _Float16* abase = xh  + (size_t)row0 * HIDN + h * HS;
  const _Float16* cbase = xch + (size_t)row0 * HIDN + h * HS;
  const _Float16* hbase = hh  + (size_t)row0 * HIDN + h * HS;
#pragma unroll
  for (int ks = 0; ks < 4; ++ks) {
    ax[ks] = load_tile_mk(abase + ks * 32, HIDN);
    ac[ks] = load_tile_mk(cbase + ks * 32, HIDN);
    ah[ks] = load_tile_mk(hbase + ks * 32, HIDN);
  }

  for (int nt = 0; nt < 8; ++nt) {
    const int n0 = nt * 16;
    const size_t wb = ((size_t)h * HS + n0) * HS;  // [h][n][k] image, ld = HS
    v8f az = {}, ai = {}, af = {}, ao = {};
#pragma unroll
    for (int ks = 0; ks < 4; ++ks) {
      const int k0 = ks * 32;
      az = wmma_f16(ax[ks], load_tile_mk(wzT + wb + k0, HS), az);
      az = wmma_f16(ah[ks], load_tile_mk(rzT + wb + k0, HS), az);
      ao = wmma_f16(ax[ks], load_tile_mk(woT + wb + k0, HS), ao);
      ao = wmma_f16(ah[ks], load_tile_mk(roT + wb + k0, HS), ao);
      ai = wmma_f16(ac[ks], load_tile_mk(wiT + wb + k0, HS), ai);
      ai = wmma_f16(ah[ks], load_tile_mk(riT + wb + k0, HS), ai);
      af = wmma_f16(ac[ks], load_tile_mk(wfT + wb + k0, HS), af);
      af = wmma_f16(ah[ks], load_tile_mk(rfT + wb + k0, HS), af);
    }
    const int gc = h * HS + n0 + nlane;        // global hidden column
    const float zb = wz_b[gc] + rz_b[gc];
    const float ib = wi_b[gc] + ri_b[gc];
    const float fb = wf_b[gc] + rf_b[gc];
    const float ob = wo_b[gc] + ro_b[gc];
#pragma unroll
    for (int v = 0; v < 8; ++v) {
      const int lr = v + 8 * rsel;             // local row (C/D layout)
      const size_t gi = (size_t)(row0 + lr) * HIDN + gc;
      const float zt = tanhf(az[v] + zb);
      const float ot = 1.f / (1.f + expf(-(ao[v] + ob)));
      const float it = ai[v] + ib;
      const float ft = af[v] + fb;
      const float mp = m_prev[gi];
      const float mt = fmaxf(ft + mp, it);
      const float ig = expf(it - mt);
      const float fg = expf(ft + mp - mt);
      const float ct = fg * c_prev[gi] + ig * zt;
      const float nv = fg * n_prev[gi] + ig;
      const float ht = ot * ct / nv;
      out_m[gi] = mt; out_c[gi] = ct; out_n[gi] = nv; out_h[gi] = ht;
      ht_s[lr][gc] = (_Float16)ht;
    }
  }
  __syncthreads();

  // GroupNorm stats: lane r (<16) reduces its row over this head's 128 cols.
  if (lane < 16) {
    float s = 0.f, s2 = 0.f;
#pragma unroll 4
    for (int c = 0; c < HS; ++c) {
      const float v = (float)ht_s[lane][h * HS + c];
      s += v; s2 += v * v;
    }
    const float mu  = s * (1.0f / HS);
    const float var = s2 * (1.0f / HS) - mu * mu;
    mu_s[h][lane] = mu;
    iv_s[h][lane] = rsqrtf(var + 1e-5f);
  }
  __syncthreads();

  for (int t = lane; t < 16 * HS; t += 32) {
    const int r  = t >> 7;
    const int c  = t & (HS - 1);
    const int gc = h * HS + c;
    const float v =
        ((float)ht_s[r][gc] - mu_s[h][r]) * iv_s[h][r] * gn_w[gc] + gn_b[gc];
    onorm[(size_t)(row0 + r) * HIDN + gc] = (_Float16)v;
  }
}

// ---------------------------------------------------------------------------
// Kernel 4: up-projections + gated GeLU. One wave computes a 64x16 output
// tile for BOTH left and right projections: per K-step, 2 B fragments are
// shared by 4 M-subtiles (8 v_wmma per 6 fragment loads) -> 4x less weight
// traffic than 16x16 blocking. K = 1024 in 32-steps.
// ---------------------------------------------------------------------------
__global__ __launch_bounds__(32) void k_up(
    const _Float16* __restrict__ onorm, const _Float16* __restrict__ upLT,
    const _Float16* __restrict__ upRT, const float* __restrict__ upL_b,
    const float* __restrict__ upR_b, _Float16* __restrict__ g) {
  const int m0 = blockIdx.x * 64;
  const int n0 = blockIdx.y * 16;
  v8f accL[4] = {};
  v8f accR[4] = {};
  const _Float16* a0  = onorm + (size_t)m0 * HIDN;
  const _Float16* bl0 = upLT  + (size_t)n0 * HIDN;
  const _Float16* br0 = upRT  + (size_t)n0 * HIDN;
  for (int k0 = 0; k0 < HIDN; k0 += 32) {
    const v16h bl = load_tile_mk(bl0 + k0, HIDN);
    const v16h br = load_tile_mk(br0 + k0, HIDN);
#pragma unroll
    for (int mt = 0; mt < 4; ++mt) {
      const v16h a = load_tile_mk(a0 + (size_t)mt * 16 * HIDN + k0, HIDN);
      accL[mt] = wmma_f16(a, bl, accL[mt]);
      accR[mt] = wmma_f16(a, br, accR[mt]);
    }
  }
  const int lane = threadIdx.x;
  const int col  = n0 + (lane & 15);
  const int rsel = lane >> 4;
  const float bL = (col < PDIM) ? upL_b[col] : 0.f;
  const float bR = (col < PDIM) ? upR_b[col] : 0.f;
#pragma unroll
  for (int mt = 0; mt < 4; ++mt) {
#pragma unroll
    for (int v = 0; v < 8; ++v) {
      const int row = m0 + mt * 16 + v + 8 * rsel;
      const float l = accL[mt][v] + bL;
      const float r = accR[mt][v] + bR;
      const float ge = 0.5f * r * (1.f + erff(r * 0.70710678118654752f));
      g[(size_t)row * PPAD + col] = (_Float16)(l * ge);  // pad cols -> exact 0
    }
  }
}

// ---------------------------------------------------------------------------
// Kernel 5: down-projection + bias + residual. 64x16 tile per wave:
// one B fragment feeds 4 M-subtiles per K-step. K = 1376 (zero padded).
// ---------------------------------------------------------------------------
__global__ __launch_bounds__(32) void k_down(
    const _Float16* __restrict__ g, const _Float16* __restrict__ downT,
    const float* __restrict__ down_b, const float* __restrict__ x,
    float* __restrict__ out) {
  const int m0 = blockIdx.x * 64;
  const int n0 = blockIdx.y * 16;
  v8f acc[4] = {};
  const _Float16* a0 = g     + (size_t)m0 * PPAD;
  const _Float16* b0 = downT + (size_t)n0 * PPAD;
  for (int k0 = 0; k0 < PPAD; k0 += 32) {
    const v16h b = load_tile_mk(b0 + k0, PPAD);
#pragma unroll
    for (int mt = 0; mt < 4; ++mt) {
      const v16h a = load_tile_mk(a0 + (size_t)mt * 16 * PPAD + k0, PPAD);
      acc[mt] = wmma_f16(a, b, acc[mt]);
    }
  }
  const int lane = threadIdx.x;
  const int col  = n0 + (lane & 15);
  const int rsel = lane >> 4;
  const float bb = down_b[col];
#pragma unroll
  for (int mt = 0; mt < 4; ++mt) {
#pragma unroll
    for (int v = 0; v < 8; ++v) {
      const int row = m0 + mt * 16 + v + 8 * rsel;
      const size_t gi = (size_t)row * DIM + col;
      out[gi] = acc[mt][v] + bb + x[gi];
    }
  }
}

// ---------------------------------------------------------------------------
// Host launch.
// ---------------------------------------------------------------------------
extern "C" void kernel_launch(void* const* d_in, const int* in_sizes, int n_in,
                              void* d_out, int out_size, void* d_ws,
                              size_t ws_size, hipStream_t stream) {
  (void)in_sizes; (void)n_in; (void)out_size; (void)ws_size;

  // Inputs (setup_inputs order)
  const float* x      = (const float*)d_in[0];
  const float* h_prev = (const float*)d_in[1];
  const float* c_prev = (const float*)d_in[2];
  const float* n_prev = (const float*)d_in[3];
  const float* m_prev = (const float*)d_in[4];
  const float* rms_w  = (const float*)d_in[5];
  const float* conv_w = (const float*)d_in[6];
  const float* conv_b = (const float*)d_in[7];
  const float* Wz_w = (const float*)d_in[8];  const float* Wz_b = (const float*)d_in[9];
  const float* Wi_w = (const float*)d_in[10]; const float* Wi_b = (const float*)d_in[11];
  const float* Wf_w = (const float*)d_in[12]; const float* Wf_b = (const float*)d_in[13];
  const float* Wo_w = (const float*)d_in[14]; const float* Wo_b = (const float*)d_in[15];
  const float* Rz_w = (const float*)d_in[16]; const float* Rz_b = (const float*)d_in[17];
  const float* Ri_w = (const float*)d_in[18]; const float* Ri_b = (const float*)d_in[19];
  const float* Rf_w = (const float*)d_in[20]; const float* Rf_b = (const float*)d_in[21];
  const float* Ro_w = (const float*)d_in[22]; const float* Ro_b = (const float*)d_in[23];
  const float* gn_w = (const float*)d_in[24]; const float* gn_b = (const float*)d_in[25];
  const float* upL_w = (const float*)d_in[26]; const float* upL_b = (const float*)d_in[27];
  const float* upR_w = (const float*)d_in[28]; const float* upR_b = (const float*)d_in[29];
  const float* down_w = (const float*)d_in[30]; const float* down_b = (const float*)d_in[31];

  // Outputs: (final, h_t, c_t, n_t, m_t) concatenated, fp32.
  float* out_final = (float*)d_out;
  float* out_h = out_final + (size_t)BATCH * DIM;
  float* out_c = out_h + (size_t)BATCH * HIDN;
  float* out_n = out_c + (size_t)BATCH * HIDN;
  float* out_m = out_n + (size_t)BATCH * HIDN;

  // Workspace carve-up (f16 staging + transposed f16 weights), 256B aligned.
  char* ws = (char*)d_ws;
  size_t off = 0;
  auto carve = [&](size_t bytes) -> char* {
    char* p = ws + off;
    off = (off + bytes + 255) & ~(size_t)255;
    return p;
  };
  const size_t actH = (size_t)BATCH * HIDN * sizeof(_Float16);
  _Float16* xh    = (_Float16*)carve(actH);
  _Float16* xch   = (_Float16*)carve(actH);
  _Float16* hh    = (_Float16*)carve(actH);
  _Float16* onorm = (_Float16*)carve(actH);
  _Float16* gbuf  = (_Float16*)carve((size_t)BATCH * PPAD * sizeof(_Float16));
  const size_t gateW = (size_t)NH * HS * HS * sizeof(_Float16);
  _Float16* wzT = (_Float16*)carve(gateW);
  _Float16* wiT = (_Float16*)carve(gateW);
  _Float16* wfT = (_Float16*)carve(gateW);
  _Float16* woT = (_Float16*)carve(gateW);
  _Float16* rzT = (_Float16*)carve(gateW);
  _Float16* riT = (_Float16*)carve(gateW);
  _Float16* rfT = (_Float16*)carve(gateW);
  _Float16* roT = (_Float16*)carve(gateW);
  _Float16* upLT  = (_Float16*)carve((size_t)PPAD * HIDN * sizeof(_Float16));
  _Float16* upRT  = (_Float16*)carve((size_t)PPAD * HIDN * sizeof(_Float16));
  _Float16* downT = (_Float16*)carve((size_t)DIM * PPAD * sizeof(_Float16));

  // 1) Activation prep.
  k_prep<<<BATCH, 256, 0, stream>>>(x, h_prev, rms_w, conv_w, conv_b,
                                    xh, xch, hh);

  // 2) Weight convert + transpose. Gates: [8][128][128] -> [8][N=128][K=128].
  {
    const int tot = HS * HS;
    dim3 gg((tot + 255) / 256, 1, NH);
    k_wt<<<gg, 256, 0, stream>>>(Wz_w, wzT, HS, HS, HS, HS);
    k_wt<<<gg, 256, 0, stream>>>(Wi_w, wiT, HS, HS, HS, HS);
    k_wt<<<gg, 256, 0, stream>>>(Wf_w, wfT, HS, HS, HS, HS);
    k_wt<<<gg, 256, 0, stream>>>(Wo_w, woT, HS, HS, HS, HS);
    k_wt<<<gg, 256, 0, stream>>>(Rz_w, rzT, HS, HS, HS, HS);
    k_wt<<<gg, 256, 0, stream>>>(Ri_w, riT, HS, HS, HS, HS);
    k_wt<<<gg, 256, 0, stream>>>(Rf_w, rfT, HS, HS, HS, HS);
    k_wt<<<gg, 256, 0, stream>>>(Ro_w, roT, HS, HS, HS, HS);
    const size_t totU = (size_t)PPAD * HIDN;
    dim3 gu((unsigned)((totU + 255) / 256), 1, 1);
    k_wt<<<gu, 256, 0, stream>>>(upL_w, upLT, HIDN, PDIM, HIDN, PPAD);
    k_wt<<<gu, 256, 0, stream>>>(upR_w, upRT, HIDN, PDIM, HIDN, PPAD);
    const size_t totD = (size_t)DIM * PPAD;
    dim3 gd((unsigned)((totD + 255) / 256), 1, 1);
    k_wt<<<gd, 256, 0, stream>>>(down_w, downT, PDIM, DIM, PPAD, DIM);
  }

  // 3) Gates + cell + GroupNorm.
  k_cell<<<BATCH / 16, dim3(32, NH), 0, stream>>>(
      xh, xch, hh, wzT, wiT, wfT, woT, rzT, riT, rfT, roT,
      Wz_b, Wi_b, Wf_b, Wo_b, Rz_b, Ri_b, Rf_b, Ro_b,
      c_prev, n_prev, m_prev, gn_w, gn_b,
      out_h, out_c, out_n, out_m, onorm);

  // 4) Up projections + gated GeLU (64x16 tiles per wave).
  k_up<<<dim3(BATCH / 64, PPAD / 16), 32, 0, stream>>>(
      onorm, upLT, upRT, upL_b, upR_b, gbuf);

  // 5) Down projection + residual (64x16 tiles per wave).
  k_down<<<dim3(BATCH / 64, DIM / 16), 32, 0, stream>>>(
      gbuf, downT, down_b, x, out_final);
}
